// HierarchicalMoELayer_9234179687030
// MI455X (gfx1250) — compile-verified, matrix operand order checked by reading, not run on hardware
//
#include <hip/hip_runtime.h>
#include <math.h>

#define D_MODEL 512
#define D_FF    2048
#define NEXP    64
#define CAPCITY 1024
#define TM      64        // token rows per workgroup
#define FC      64        // D_FF chunk width
#define NTOK    8192      // B*S
#define NASSIGN 16384     // NTOK * TOPK
#define SMEM_BYTES (TM*D_MODEL*2 + FC*D_MODEL*2 + D_MODEL*FC*2 + TM*FC*2 + TM*4 + TM*4)

typedef __attribute__((ext_vector_type(16))) __bf16 v16bf;
typedef __attribute__((ext_vector_type(8)))  __bf16 v8bf;
typedef __attribute__((ext_vector_type(4)))  __bf16 v4bf;
typedef __attribute__((ext_vector_type(8)))  float  v8f;

// Load a 16x32 bf16 A-fragment (or symmetric 32x16 B-fragment, row = N of B^T)
// per CDNA5 ISA layout: lane r=lane&15 holds row r; kbase=(lane>>4)*8;
// halves 0..7 -> K=kbase..kbase+7, halves 8..15 -> K=kbase+16..kbase+23.
static __device__ __forceinline__ v16bf ld_frag(const __bf16* base, int stride, int lane) {
  const int r  = lane & 15;
  const int kb = (lane >> 4) << 3;
  const __bf16* p = base + r * stride + kb;
  v8bf lo = *(const v8bf*)(p);
  v8bf hi = *(const v8bf*)(p + 16);
  v16bf o;
#pragma unroll
  for (int i = 0; i < 8; ++i) { o[i] = lo[i]; o[i + 8] = hi[i]; }
  return o;
}

static __device__ __forceinline__ v8f zero8() {
  v8f z;
#pragma unroll
  for (int i = 0; i < 8; ++i) z[i] = 0.0f;
  return z;
}

static __device__ __forceinline__ float gelu_exact(float x) {
  return 0.5f * x * (1.0f + erff(x * 0.70710678118654752f));
}

__global__ void moe_zero_kernel(float* __restrict__ out, int* __restrict__ counts) {
  int i = blockIdx.x * blockDim.x + threadIdx.x;
  if (i < NTOK * D_MODEL) out[i] = 0.0f;
  if (i < NEXP) counts[i] = 0;
}

__global__ void moe_route_kernel(const int* __restrict__ cat, const int* __restrict__ sub,
                                 const float* __restrict__ wts,
                                 int* __restrict__ counts, int* __restrict__ tokbuf,
                                 float* __restrict__ wgtbuf) {
  int a = blockIdx.x * blockDim.x + threadIdx.x;
  if (a >= NASSIGN) return;
  int e = cat[a] * 8 + sub[a];
  int pos = atomicAdd(counts + e, 1);
  if (pos < CAPCITY) {
    tokbuf[e * CAPCITY + pos] = a >> 1;   // token = assignment / TOPK (TOPK=2)
    wgtbuf[e * CAPCITY + pos] = wts[a];
  }
}

// One workgroup = one expert x 64-token tile. 8 waves, each owns 64 output cols.
__global__ void __launch_bounds__(256)
moe_expert_mlp_kernel(const float* __restrict__ hidden,
                      const float* __restrict__ W1g, const float* __restrict__ b1g,
                      const float* __restrict__ W2g, const float* __restrict__ b2g,
                      const int*  __restrict__ counts, const int* __restrict__ tokbuf,
                      const float* __restrict__ wgtbuf, float* __restrict__ out) {
  extern __shared__ char smem[];
  __bf16* Xl   = (__bf16*)smem;                 // [TM][512]   X tile (bf16)
  __bf16* W1l  = Xl  + TM * D_MODEL;            // [FC f][512 k]  W1 chunk, transposed
  __bf16* W2l  = W1l + FC * D_MODEL;            // [512 n][FC k]  W2 chunk, transposed
  __bf16* Hl   = W2l + D_MODEL * FC;            // [TM][FC]    GELU intermediate (bf16)
  int*    TokL = (int*)(Hl + TM * FC);          // [TM]
  float*  WgtL = (float*)(TokL + TM);           // [TM]

  const int e  = blockIdx.x >> 4;               // 64 experts
  const int m0 = (blockIdx.x & 15) * TM;        // tile start within capacity
  int count = counts[e];
  if (count > CAPCITY) count = CAPCITY;
  if (m0 >= count) return;

  const int tid  = threadIdx.x;
  const int wave = tid >> 5;
  const int lane = tid & 31;

  if (tid < TM) {
    int mg = m0 + tid;
    bool v = mg < count;
    TokL[tid] = v ? tokbuf[e * CAPCITY + mg] : 0;
    WgtL[tid] = v ? wgtbuf[e * CAPCITY + mg] : 0.0f;
  }
  __syncthreads();

  { // stage X tile: gather routed token rows, f32 -> bf16
    int row = tid >> 2;
    int cb  = (tid & 3) * 128;
    bool v  = (m0 + row) < count;
    int tk  = TokL[row];
    const float4* src = (const float4*)(hidden + (size_t)tk * D_MODEL + cb);
#pragma unroll
    for (int i = 0; i < 32; ++i) {
      float4 f = v ? src[i] : make_float4(0.f, 0.f, 0.f, 0.f);
      v4bf b; b[0] = (__bf16)f.x; b[1] = (__bf16)f.y; b[2] = (__bf16)f.z; b[3] = (__bf16)f.w;
      *(v4bf*)(Xl + row * D_MODEL + cb + i * 4) = b;
    }
  }

  v8f Yacc[4][4];
#pragma unroll
  for (int a = 0; a < 4; ++a)
#pragma unroll
    for (int b = 0; b < 4; ++b) Yacc[a][b] = zero8();

  for (int f0 = 0; f0 < D_FF; f0 += FC) {
    __syncthreads();  // prev stage2 done before overwriting W1l/W2l/Hl

    // stage W1[:, f0:f0+64] transposed -> W1l[f][k]
#pragma unroll
    for (int kk = 0; kk < 2; ++kk) {
      int k = tid + kk * 256;
      const float4* src = (const float4*)(W1g + ((size_t)e * D_MODEL + k) * D_FF + f0);
#pragma unroll
      for (int i = 0; i < 16; ++i) {
        float4 f = src[i];
        int fl = i * 4;
        W1l[(fl + 0) * D_MODEL + k] = (__bf16)f.x;
        W1l[(fl + 1) * D_MODEL + k] = (__bf16)f.y;
        W1l[(fl + 2) * D_MODEL + k] = (__bf16)f.z;
        W1l[(fl + 3) * D_MODEL + k] = (__bf16)f.w;
      }
    }
    // stage W2[f0:f0+64, :] transposed -> W2l[n][k]
    {
      int kl = tid >> 2;
      int nb = (tid & 3) * 128;
      const float4* src = (const float4*)(W2g + ((size_t)e * D_FF + f0 + kl) * D_MODEL + nb);
#pragma unroll
      for (int i = 0; i < 32; ++i) {
        float4 f = src[i];
        int n = nb + i * 4;
        W2l[(n + 0) * FC + kl] = (__bf16)f.x;
        W2l[(n + 1) * FC + kl] = (__bf16)f.y;
        W2l[(n + 2) * FC + kl] = (__bf16)f.z;
        W2l[(n + 3) * FC + kl] = (__bf16)f.w;
      }
    }
    if (f0 + FC < D_FF) {  // global_prefetch_b8 next chunk
      __builtin_prefetch(W1g + ((size_t)e * D_MODEL + tid) * D_FF + f0 + FC, 0, 1);
      __builtin_prefetch(W2g + ((size_t)e * D_FF + f0 + FC + (tid >> 2)) * D_MODEL, 0, 1);
    }
    __syncthreads();

    // ---- stage 1: H = GELU(X @ W1chunk + b1) ; 16 tiles over 8 waves ----
#pragma unroll
    for (int tt = 0; tt < 2; ++tt) {
      int t2 = wave * 2 + tt;
      int mi = t2 >> 2, ni = t2 & 3;
      v8f acc = zero8();
#pragma unroll
      for (int ks = 0; ks < D_MODEL / 32; ++ks) {
        v16bf a = ld_frag(Xl  + mi * 16 * D_MODEL + ks * 32, D_MODEL, lane);
        v16bf b = ld_frag(W1l + ni * 16 * D_MODEL + ks * 32, D_MODEL, lane);
        acc = __builtin_amdgcn_wmma_f32_16x16x32_bf16(false, a, false, b, (short)0, acc, false, false);
      }
      float b1v = b1g[(size_t)e * D_FF + f0 + ni * 16 + (lane & 15)];
      int mb = mi * 16 + ((lane >> 4) & 1) * 8;
      int nc = ni * 16 + (lane & 15);
#pragma unroll
      for (int r = 0; r < 8; ++r) {
        float g = gelu_exact(acc[r] + b1v);
        Hl[(mb + r) * FC + nc] = (__bf16)g;
      }
    }
    __syncthreads();

    // ---- stage 2: Yacc += H @ W2chunk ; wave owns cols [wave*64, wave*64+64) ----
#pragma unroll
    for (int mi = 0; mi < 4; ++mi) {
#pragma unroll
      for (int kk = 0; kk < 2; ++kk) {
        v16bf a = ld_frag(Hl + mi * 16 * FC + kk * 32, FC, lane);
#pragma unroll
        for (int ni = 0; ni < 4; ++ni) {
          v16bf b = ld_frag(W2l + (wave * 64 + ni * 16) * FC + kk * 32, FC, lane);
          Yacc[mi][ni] = __builtin_amdgcn_wmma_f32_16x16x32_bf16(false, a, false, b, (short)0, Yacc[mi][ni], false, false);
        }
      }
    }
  }

  // ---- epilogue: out[token] += weight * (Y + b2), scatter-add over tokens ----
#pragma unroll
  for (int mi = 0; mi < 4; ++mi) {
#pragma unroll
    for (int ni = 0; ni < 4; ++ni) {
      int n = wave * 64 + ni * 16 + (lane & 15);
      float b2v = b2g[(size_t)e * D_MODEL + n];
      int mloc = mi * 16 + ((lane >> 4) & 1) * 8;
#pragma unroll
      for (int r = 0; r < 8; ++r) {
        int m = mloc + r;
        if (m0 + m < count) {
          float val = Yacc[mi][ni][r] + b2v;
          atomicAdd(out + (size_t)TokL[m] * D_MODEL + n, WgtL[m] * val);
        }
      }
    }
  }
}

extern "C" void kernel_launch(void* const* d_in, const int* in_sizes, int n_in,
                              void* d_out, int out_size, void* d_ws, size_t ws_size,
                              hipStream_t stream) {
  const float* hidden = (const float*)d_in[0];
  const int*   cat    = (const int*)d_in[1];
  const int*   sub    = (const int*)d_in[2];
  const float* wts    = (const float*)d_in[3];
  const float* W1     = (const float*)d_in[4];
  const float* b1     = (const float*)d_in[5];
  const float* W2     = (const float*)d_in[6];
  const float* b2     = (const float*)d_in[7];
  float* out = (float*)d_out;

  int*   counts = (int*)d_ws;                       // 64
  int*   tokbuf = counts + 64;                      // 64*1024
  float* wgtbuf = (float*)(tokbuf + NEXP * CAPCITY);// 64*1024

  (void)hipFuncSetAttribute((const void*)moe_expert_mlp_kernel,
                            hipFuncAttributeMaxDynamicSharedMemorySize, SMEM_BYTES);

  moe_zero_kernel<<<(NTOK * D_MODEL + 255) / 256, 256, 0, stream>>>(out, counts);
  moe_route_kernel<<<NASSIGN / 256, 256, 0, stream>>>(cat, sub, wts, counts, tokbuf, wgtbuf);
  moe_expert_mlp_kernel<<<NEXP * (CAPCITY / TM), 256, SMEM_BYTES, stream>>>(
      hidden, W1, b1, W2, b2, counts, tokbuf, wgtbuf, out);
}
